// VectorQuantizer_72619307040970
// MI455X (gfx1250) — matrix operand order repaired and checked by reading, not compiled, not run on hardware
//
#include <hip/hip_runtime.h>
#include <hip/hip_bf16.h>
#include <stdint.h>

// ---------------- problem constants ----------------
#define N_TOKENS 131072
#define N_E      1024
#define E_DIM    256
#define BETA     0.25f
#define GAMMA    0.99f

// ---------------- output layout (flat floats, tuple return order) ----------
// quant, commit, z_q_st(33554432), perplexity, one_hot(134217728),
// idx(131072), new_emb(262144), new_n(1024), new_m(262144)
static const size_t OFS_QUANT  = 0;
static const size_t OFS_COMMIT = 1;
static const size_t OFS_ZQ     = 2;
static const size_t OFS_PERP   = 2 + (size_t)N_TOKENS * E_DIM;            // 33554434
static const size_t OFS_ONEHOT = OFS_PERP + 1;                            // 33554435
static const size_t OFS_IDX    = OFS_ONEHOT + (size_t)N_TOKENS * N_E;     // 167772163
static const size_t OFS_EMB    = OFS_IDX + (size_t)N_TOKENS;              // 167903235
static const size_t OFS_N      = OFS_EMB + (size_t)N_E * E_DIM;           // 168165379
static const size_t OFS_M      = OFS_N + (size_t)N_E;                     // 168166403

// ---------------- workspace layout (bytes) ----------------
// [0)              embh   : 262144 * 2B  (f16 copy of codebook, row-major NxK)
// [524288)         enorm  : 1024 * 4B    (||e_j||^2, f32)
// [528384)         idx    : 131072 * 4B  (int32 argmin)
// [1052672)        accum  : counts(1024) + sums(262144) + sse(1) f32, zeroed per call
static const size_t WSB_EMBH  = 0;
static const size_t WSB_ENORM = 524288;
static const size_t WSB_IDX   = 528384;
static const size_t WSB_ACCUM = 1052672;
static const size_t N_ACCUM   = (size_t)N_E + (size_t)N_E * E_DIM + 1;    // 263169 floats

typedef _Float16 v16h __attribute__((ext_vector_type(16)));
typedef _Float16 v8h  __attribute__((ext_vector_type(8)));
typedef float    v8f  __attribute__((ext_vector_type(8)));

// ============================================================
// 0) zero the atomic accumulators (counts, sums, sse)
// ============================================================
__global__ void vq_zero_kernel(float* __restrict__ accum, size_t n) {
    size_t i = (size_t)blockIdx.x * blockDim.x + threadIdx.x;
    size_t stride = (size_t)gridDim.x * blockDim.x;
    for (; i < n; i += stride) accum[i] = 0.0f;
}

// ============================================================
// 1) codebook prep: f32 -> f16 copy + ||e_j||^2   (1024 blocks x 256 thr)
// ============================================================
__global__ void vq_prep_kernel(const float* __restrict__ emb,
                               _Float16* __restrict__ embh,
                               float* __restrict__ enorm) {
    const int j = blockIdx.x;       // codeword
    const int d = threadIdx.x;      // dim
    float w = emb[j * E_DIM + d];
    embh[j * E_DIM + d] = (_Float16)w;
    float s = w * w;
    #pragma unroll
    for (int m = 16; m >= 1; m >>= 1) s += __shfl_xor(s, m, 32);
    __shared__ float partial[8];
    if ((d & 31) == 0) partial[d >> 5] = s;
    __syncthreads();
    if (d == 0) {
        float t = 0.f;
        #pragma unroll
        for (int i = 0; i < 8; ++i) t += partial[i];
        enorm[j] = t;
    }
}

// ============================================================
// 2) WMMA nearest-codeword search.
//    Block: 128 threads (4 waves), 64 tokens (16 rows/wave).
//    argmin_j d = argmax_j (z.e_j - 0.5*||e_j||^2)  -> v_wmma_f32_16x16x32_f16
// ============================================================
#define TOKS_PER_BLK 64
#define ZPITCH 264   // halves per row; 528B row stride -> conflict-free & 16B aligned

__global__ __launch_bounds__(128)
void vq_argmin_kernel(const float* __restrict__ z,
                      const _Float16* __restrict__ embh,
                      const float* __restrict__ enorm,
                      int* __restrict__ idx_out) {
    __shared__ _Float16 zsh[TOKS_PER_BLK * ZPITCH];

    const int tid  = threadIdx.x;
    const int lane = tid & 31;
    const int wave = tid >> 5;
    const size_t blockRow = (size_t)blockIdx.x * TOKS_PER_BLK;

    // cooperative stage: 64x256 f32 -> f16 in LDS (float4 global loads)
    for (int i = tid; i < TOKS_PER_BLK * (E_DIM / 4); i += 128) {
        const int row = i >> 6;            // 64 float4 per row
        const int c4  = (i & 63) << 2;
        const float4 v = *reinterpret_cast<const float4*>(z + (blockRow + row) * E_DIM + c4);
        _Float16* dst = &zsh[row * ZPITCH + c4];
        dst[0] = (_Float16)v.x; dst[1] = (_Float16)v.y;
        dst[2] = (_Float16)v.z; dst[3] = (_Float16)v.w;
    }
    __syncthreads();

    const int mrow    = wave * 16;
    const int halfSel = lane >> 4;     // 0: lanes 0-15, 1: lanes 16-31
    const int nLane   = lane & 15;
    // A-matrix 16x32 f16 layout: lane holds M=lane%16, K-chunks [a..a+7] and [a+16..a+23], a=8*halfSel
    const _Float16* arow = &zsh[(mrow + nLane) * ZPITCH + halfSel * 8];

    float bestV[8];
    int   bestI[8];
    #pragma unroll
    for (int r = 0; r < 8; ++r) { bestV[r] = -3.0e38f; bestI[r] = 0; }

    for (int nt = 0; nt < N_E / 16; ++nt) {
        const int col = nt * 16 + nLane;
        const float en = enorm[col];
        // B-matrix 32x16 f16 layout: lane holds N=lane%16, K-chunks [a..a+7] and [a+16..a+23], a=8*halfSel
        const _Float16* brow = embh + (size_t)col * E_DIM + halfSel * 8;

        v8f c = {0.f, 0.f, 0.f, 0.f, 0.f, 0.f, 0.f, 0.f};
        #pragma unroll
        for (int k = 0; k < E_DIM / 32; ++k) {      // 8 K-steps of 32
            const int kb = k * 32;
            v8h alo = *reinterpret_cast<const v8h*>(arow + kb);
            v8h ahi = *reinterpret_cast<const v8h*>(arow + kb + 16);
            v16h a = __builtin_shufflevector(alo, ahi,
                     0,1,2,3,4,5,6,7,8,9,10,11,12,13,14,15);
            v8h blo = *reinterpret_cast<const v8h*>(brow + kb);
            v8h bhi = *reinterpret_cast<const v8h*>(brow + kb + 16);
            v16h b = __builtin_shufflevector(blo, bhi,
                     0,1,2,3,4,5,6,7,8,9,10,11,12,13,14,15);
            c = __builtin_amdgcn_wmma_f32_16x16x32_f16(
                    false, a, false, b, (short)0, c, false, false);
        }
        // C/D layout: VGPR r, lane -> row M = r + 8*halfSel, col N = nt*16 + nLane
        #pragma unroll
        for (int r = 0; r < 8; ++r) {
            const float val = c[r] - 0.5f * en;
            if (val > bestV[r]) { bestV[r] = val; bestI[r] = col; }  // strict: earliest wins
        }
    }

    // per-row reduction across the 16-lane group (same M per group); tie -> lower index
    #pragma unroll
    for (int r = 0; r < 8; ++r) {
        float v = bestV[r];
        int  ix = bestI[r];
        #pragma unroll
        for (int m = 8; m >= 1; m >>= 1) {
            float ov = __shfl_xor(v, m, 16);
            int   oi = __shfl_xor(ix, m, 16);
            if (ov > v || (ov == v && oi < ix)) { v = ov; ix = oi; }
        }
        if (nLane == 0)
            idx_out[blockRow + mrow + r + 8 * halfSel] = ix;
    }
}

// ============================================================
// 3) gather/scatter: z_q_st, segment sums (f32 atomics), counts, SSE, idx->float
//    one block per token, 256 threads (one per dim)
// ============================================================
__global__ void vq_scatter_kernel(const float* __restrict__ z,
                                  const float* __restrict__ emb,
                                  const int* __restrict__ idx,
                                  float* __restrict__ out,
                                  float* __restrict__ counts,
                                  float* __restrict__ sums,
                                  float* __restrict__ sse) {
    const size_t t = blockIdx.x;
    const int d = threadIdx.x;
    const int j = idx[t];
    const float zv = z[t * E_DIM + d];
    const float ev = emb[(size_t)j * E_DIM + d];
    const float diff = ev - zv;                       // z_q - z
    __builtin_nontemporal_store(zv + diff, out + OFS_ZQ + t * E_DIM + d); // straight-through
    atomicAdd(&sums[(size_t)j * E_DIM + d], zv);      // segment_sum
    float sq = diff * diff;
    #pragma unroll
    for (int m = 16; m >= 1; m >>= 1) sq += __shfl_xor(sq, m, 32);
    __shared__ float partial[8];
    if ((d & 31) == 0) partial[d >> 5] = sq;
    __syncthreads();
    if (d == 0) {
        float s = 0.f;
        #pragma unroll
        for (int i = 0; i < 8; ++i) s += partial[i];
        atomicAdd(sse, s);
        atomicAdd(&counts[j], 1.0f);
        out[OFS_IDX + t] = (float)j;
    }
}

// ============================================================
// 4) one-hot encodings (537 MB, nontemporal): 4 cols per thread
// ============================================================
__global__ void vq_onehot_kernel(const int* __restrict__ idx,
                                 float* __restrict__ out) {
    const size_t i = ((size_t)blockIdx.x * blockDim.x + threadIdx.x) * 4;
    const size_t row = i >> 10;            // N_E == 1024
    const int c = (int)(i & 1023);
    const int j = idx[row];
    float* o = out + OFS_ONEHOT + i;
    __builtin_nontemporal_store((c     == j) ? 1.0f : 0.0f, o + 0);
    __builtin_nontemporal_store((c + 1 == j) ? 1.0f : 0.0f, o + 1);
    __builtin_nontemporal_store((c + 2 == j) ? 1.0f : 0.0f, o + 2);
    __builtin_nontemporal_store((c + 3 == j) ? 1.0f : 0.0f, o + 3);
}

// ============================================================
// 5) EMA update: new_n, new_m, new_emb
// ============================================================
__global__ void vq_update_kernel(const float* __restrict__ n_mat,
                                 const float* __restrict__ m_mat,
                                 const float* __restrict__ counts,
                                 const float* __restrict__ sums,
                                 float* __restrict__ out) {
    const int i = blockIdx.x * 256 + threadIdx.x;     // 0..262143
    const int j = i >> 8;
    const float nm = n_mat[j] * GAMMA + counts[j] * (1.0f - GAMMA);
    const float mm = GAMMA * m_mat[i] + (1.0f - GAMMA) * sums[i];
    out[OFS_M + i]   = mm;
    out[OFS_EMB + i] = mm / nm;
    if ((i & 255) == 0) out[OFS_N + j] = nm;
}

// ============================================================
// 6) perplexity + losses (1 block x 1024)
// ============================================================
__global__ void vq_final_kernel(const float* __restrict__ counts,
                                const float* __restrict__ sse,
                                float* __restrict__ out) {
    const int j = threadIdx.x;
    const float em = counts[j] * (1.0f / (float)N_TOKENS);
    float t = em * logf(em + 1e-10f);
    #pragma unroll
    for (int m = 16; m >= 1; m >>= 1) t += __shfl_xor(t, m, 32);
    __shared__ float partial[32];
    if ((j & 31) == 0) partial[j >> 5] = t;
    __syncthreads();
    if (j == 0) {
        float s = 0.f;
        #pragma unroll
        for (int i = 0; i < 32; ++i) s += partial[i];
        out[OFS_PERP] = expf(-s);
        const float q = sse[0] * (1.0f / ((float)N_TOKENS * (float)E_DIM));
        out[OFS_QUANT]  = q;          // mean((z_q - z)^2)
        out[OFS_COMMIT] = BETA * q;   // beta * same mean
    }
}

// ============================================================
extern "C" void kernel_launch(void* const* d_in, const int* in_sizes, int n_in,
                              void* d_out, int out_size, void* d_ws, size_t ws_size,
                              hipStream_t stream) {
    const float* z     = (const float*)d_in[0];
    const float* emb   = (const float*)d_in[1];
    const float* n_mat = (const float*)d_in[2];
    const float* m_mat = (const float*)d_in[3];
    float* out = (float*)d_out;

    char* ws = (char*)d_ws;
    _Float16* embh  = (_Float16*)(ws + WSB_EMBH);
    float*    enorm = (float*)   (ws + WSB_ENORM);
    int*      idx   = (int*)     (ws + WSB_IDX);
    float*    accum = (float*)   (ws + WSB_ACCUM);
    float*    counts = accum;
    float*    sums   = accum + N_E;
    float*    sse    = accum + N_E + (size_t)N_E * E_DIM;

    vq_zero_kernel<<<512, 256, 0, stream>>>(accum, N_ACCUM);
    vq_prep_kernel<<<N_E, E_DIM, 0, stream>>>(emb, embh, enorm);
    vq_argmin_kernel<<<N_TOKENS / TOKS_PER_BLK, 128, 0, stream>>>(z, embh, enorm, idx);
    vq_scatter_kernel<<<N_TOKENS, E_DIM, 0, stream>>>(z, emb, idx, out, counts, sums, sse);
    vq_onehot_kernel<<<(N_TOKENS * (N_E / 4)) / 256, 256, 0, stream>>>(idx, out);
    vq_update_kernel<<<(N_E * E_DIM) / 256, 256, 0, stream>>>(n_mat, m_mat, counts, sums, out);
    vq_final_kernel<<<1, N_E, 0, stream>>>(counts, sse, out);
}
